// NeRFLoss_1322849927289
// MI455X (gfx1250) — matrix operand order repaired and checked by reading, not compiled, not run on hardware
//
#include <hip/hip_runtime.h>
#include <hip/hip_bf16.h>
#include <stdint.h>

typedef __attribute__((ext_vector_type(16))) _Float16 v16h;
typedef __attribute__((ext_vector_type(8)))  float    v8f;
typedef unsigned int u32x4 __attribute__((ext_vector_type(4)));
typedef int          i32x8 __attribute__((ext_vector_type(8)));
typedef int          i32x4 __attribute__((ext_vector_type(4)));

#define T_NEAR 2.0f
#define T_FAR  6.0f
#define NRAYS  1024
#define NC     64
#define NF     128
#define SFINE  (NC + NF)   // 192
#define CHUNK  32768       // points per MLP chunk (divides 65536 and 196608)

// Tensor Data Mover availability (this toolchain: 6-arg builtin form)
#if defined(__has_builtin)
#if __has_builtin(__builtin_amdgcn_tensor_load_to_lds) && \
    __has_builtin(__builtin_amdgcn_s_wait_tensorcnt)
#define USE_TDM 1
#endif
#endif
#ifndef USE_TDM
#define USE_TDM 0
#endif

// ---------------------------------------------------------------------------
// Weight conversion: fp32 W[K,N] (row-major) -> f16 Wt[n*Kpad + k] (transposed,
// zero-padded). Bias padded to Npad f32.
// ---------------------------------------------------------------------------
__global__ void cvt_layer(const float* __restrict__ W, const float* __restrict__ b,
                          int K, int N, int Kpad, int Npad,
                          _Float16* __restrict__ Wt, float* __restrict__ bp) {
    int i = blockIdx.x * blockDim.x + threadIdx.x;
    int total = Kpad * Npad;
    if (i >= total) return;
    int n = i / Kpad, k = i % Kpad;
    float v = (k < K && n < N) ? W[(size_t)k * N + n] : 0.0f;
    Wt[(size_t)n * Kpad + k] = (_Float16)v;
    if (k == 0) bp[n] = (n < N) ? b[n] : 0.0f;
}

// ---------------------------------------------------------------------------
// Stratified coarse samples: t = T_N + (T_F-T_N)*(s + u)/NC
// ---------------------------------------------------------------------------
__global__ void coarse_t_kernel(const float* __restrict__ u, float* __restrict__ t) {
    int i = blockIdx.x * blockDim.x + threadIdx.x;
    if (i >= NRAYS * NC) return;
    int s = i % NC;
    t[i] = T_NEAR + (T_FAR - T_NEAR) * ((float)s + u[i]) / (float)NC;
}

// ---------------------------------------------------------------------------
// Positional encoding for a chunk of points.
//   X64[row, 0:63] = pe(x, L=10) padded to 64 (f16)
//   Hr [row, 256:287] = pe(d, L=4) padded to 32 (f16)  (concat buffer for Wr)
// ---------------------------------------------------------------------------
__global__ void pe_points_kernel(const float* __restrict__ o, const float* __restrict__ dv,
                                 const float* __restrict__ t, int S, int p0, int cnt,
                                 _Float16* __restrict__ X64, _Float16* __restrict__ Hr) {
    int idx = blockIdx.x * blockDim.x + threadIdx.x;
    if (idx >= cnt) return;
    int P = p0 + idx;
    int r = P / S;
    float tt = t[P];
    _Float16* xr = X64 + (size_t)idx * 64;
    float x[3];
#pragma unroll
    for (int c = 0; c < 3; ++c) { x[c] = o[r*3+c] + tt * dv[r*3+c]; xr[c] = (_Float16)x[c]; }
#pragma unroll
    for (int c = 0; c < 3; ++c) {
        float f = 1.0f;
#pragma unroll
        for (int l = 0; l < 10; ++l) {
            float xb = x[c] * f;
            xr[3 + c*20 + l]      = (_Float16)__sinf(xb);
            xr[3 + c*20 + 10 + l] = (_Float16)__cosf(xb);
            f *= 2.0f;
        }
    }
    xr[63] = (_Float16)0.0f;
    _Float16* hr = Hr + (size_t)idx * 288 + 256;
    float dd[3];
#pragma unroll
    for (int c = 0; c < 3; ++c) { dd[c] = dv[r*3+c]; hr[c] = (_Float16)dd[c]; }
#pragma unroll
    for (int c = 0; c < 3; ++c) {
        float f = 1.0f;
#pragma unroll
        for (int l = 0; l < 4; ++l) {
            float xb = dd[c] * f;
            hr[3 + c*8 + l]     = (_Float16)__sinf(xb);
            hr[3 + c*8 + 4 + l] = (_Float16)__cosf(xb);
            f *= 2.0f;
        }
    }
#pragma unroll
    for (int k = 27; k < 32; ++k) hr[k] = (_Float16)0.0f;
}

// ---------------------------------------------------------------------------
// TDM issue: DMA a 32(K) x 64(N-rows) f16 slab of transposed weights into LDS.
// D# per CDNA5 ISA ch.8: group0 = {flags, lds_addr, global_addr, type=2},
// group1 = {data_size=2B, tensor dims/strides, tile 32x64}. 2D -> higher
// descriptor groups are zero.
// ---------------------------------------------------------------------------
#if USE_TDM
__device__ __forceinline__ void tdm_load_b_slab(const _Float16* gsrc, uint32_t lds_off,
                                                int K) {
    uint64_t ga = (uint64_t)(uintptr_t)gsrc;
    u32x4 g0;
    g0[0] = 1u;                                   // count=1 (valid), no gather/restore
    g0[1] = lds_off;                              // lds_addr [63:32]
    g0[2] = (uint32_t)(ga & 0xFFFFFFFFu);         // global_addr [95:64]
    g0[3] = (uint32_t)((ga >> 32) & 0x01FFFFFFu)  // global_addr [120:96]
          | (2u << 30);                           // type=2 ("image") [127:126]
    i32x8 g1;
    g1[0] = (int)(1u << 16);                      // data_size=1 (2 bytes)
    g1[1] = (int)(((uint32_t)K & 0xFFFFu) << 16); // tensor_dim0 lo -> [63:48]
    g1[2] = (int)(((uint32_t)K >> 16) |           // tensor_dim0 hi
                  (64u << 16));                   // tensor_dim1 lo (=64 rows)
    g1[3] = (int)(32u << 16);                     // tensor_dim1 hi=0 | tile_dim0=32
    g1[4] = (int)64;                              // tile_dim1=64 | tile_dim2=0
    g1[5] = (int)(uint32_t)K;                     // tensor_dim0_stride lo
    g1[6] = 0;                                    // stride0 hi | stride1 lo
    g1[7] = 0;                                    // stride1 hi
    i32x4 gz4 = {0, 0, 0, 0};
    i32x8 gz8 = {0, 0, 0, 0, 0, 0, 0, 0};
    __builtin_amdgcn_tensor_load_to_lds(g0, g1, gz4, gz4, gz8, 0);
}
#endif

// ---------------------------------------------------------------------------
// Block WMMA GEMM: 128 threads (4 waves) compute a 64x64 tile of
// Y = act(X @ W + b). B K-slab staged in LDS (TDM double-buffered when
// available), each wave computes 16 rows x 64 cols with one A fragment
// feeding 4 WMMAs. ACT: 0=none 1=relu.
// ---------------------------------------------------------------------------
template <int ACT>
__global__ void wmma_gemm_block_kernel(const _Float16* __restrict__ X, int lda,
                                       const _Float16* __restrict__ Wt, int K,
                                       const float* __restrict__ bias,
                                       _Float16* __restrict__ Y, int ldy) {
    __shared__ _Float16 ldsB[2][64 * 32];  // [buf][n*32 + k], 4KB each
    const int tid  = threadIdx.x;
    const int wave = tid >> 5;
    const int lane = tid & 31;
    const int hi   = lane >> 4;
    const int l15  = lane & 15;
    const int rowBase = blockIdx.x * 64;
    const int colBase = blockIdx.y * 64;
    const int m = rowBase + wave * 16 + l15;
    const uint32_t* arow = (const uint32_t*)(const void*)(X + (size_t)m * lda);

    union Frag { v16h v; uint32_t u[8]; };
    v8f acc[4];
    const v8f vzero = {0.f, 0.f, 0.f, 0.f, 0.f, 0.f, 0.f, 0.f};
#pragma unroll
    for (int c = 0; c < 4; ++c) acc[c] = vzero;

#if USE_TDM
    // prologue: wave 0 issues first slab into buffer 0
    if (tid < 32) {
        tdm_load_b_slab(Wt + (size_t)colBase * K,
                        (uint32_t)(uintptr_t)&ldsB[0][0], K);
    }
#endif

    for (int kk = 0, it = 0; kk < K; kk += 32, ++it) {
#if USE_TDM
        const int cur = it & 1;
        const bool hasNext = (kk + 32) < K;
        if (tid < 32) {
            if (hasNext) {
                tdm_load_b_slab(Wt + (size_t)colBase * K + (kk + 32),
                                (uint32_t)(uintptr_t)&ldsB[cur ^ 1][0], K);
                __builtin_amdgcn_s_wait_tensorcnt(1);
            } else {
                __builtin_amdgcn_s_wait_tensorcnt(0);
            }
        }
        __syncthreads();
#else
        const int cur = 0;
        // cooperative stage: 4KB = 256 x 16B units, 128 threads x 2
#pragma unroll
        for (int rep = 0; rep < 2; ++rep) {
            int unit = tid + rep * 128;
            int n  = unit >> 2;
            int ku = unit & 3;
            const uint4* src =
                (const uint4*)(const void*)(Wt + (size_t)(colBase + n) * K + kk) + ku;
            ((uint4*)(void*)&ldsB[0][0])[unit] = *src;
        }
        __syncthreads();
#endif
        // A fragment (16x32 f16, contiguous k = 16*hi + s, matching B)
        Frag a;
#pragma unroll
        for (int j = 0; j < 4; ++j) {
            a.u[j]     = arow[(kk + hi * 16 + 2 * j)     >> 1];
            a.u[j + 4] = arow[(kk + hi * 16 + 8 + 2 * j) >> 1];
        }
        const uint32_t* bl = (const uint32_t*)(const void*)&ldsB[cur][0];
#pragma unroll
        for (int c = 0; c < 4; ++c) {
            Frag b;
            const int nl = c * 16 + l15;
#pragma unroll
            for (int j = 0; j < 8; ++j)
                b.u[j] = bl[(nl * 32 + hi * 16 + 2 * j) >> 1];
            acc[c] = __builtin_amdgcn_wmma_f32_16x16x32_f16(
                false, a.v, false, b.v, (short)0, acc[c], false, false);
        }
        __syncthreads();
    }

    const int mB = rowBase + wave * 16 + hi * 8;
#pragma unroll
    for (int c = 0; c < 4; ++c) {
        const int n = colBase + c * 16 + l15;
        const float bv = bias[n];
#pragma unroll
        for (int i = 0; i < 8; ++i) {
            float v = acc[c][i] + bv;
            if (ACT == 1) v = fmaxf(v, 0.0f);
            Y[(size_t)(mB + i) * ldy + n] = (_Float16)v;
        }
    }
}

// ---------------------------------------------------------------------------
// Per-wave WMMA GEMM for the small heads (N=16 padded): Yf = act(X@W+b),
// f32 store of first `fcols` columns. ACT: 1=relu 2=sigmoid.
// ---------------------------------------------------------------------------
template <int ACT>
__global__ void wmma_head_kernel(const _Float16* __restrict__ X, int lda,
                                 const _Float16* __restrict__ Wt, int K,
                                 const float* __restrict__ bias,
                                 float* __restrict__ Yf, int ldyf, int fcols) {
    const int lane = threadIdx.x & 31;
    const int hi   = lane >> 4;
    const int l15  = lane & 15;
    const int rowBase = blockIdx.x * 16;
    const int m = rowBase + l15;
    const uint32_t* arow = (const uint32_t*)(const void*)(X + (size_t)m * lda);
    const uint32_t* brow = (const uint32_t*)(const void*)(Wt + (size_t)l15 * K);

    union Frag { v16h v; uint32_t u[8]; };
    v8f acc = {0.f, 0.f, 0.f, 0.f, 0.f, 0.f, 0.f, 0.f};
    for (int kk = 0; kk < K; kk += 32) {
        Frag a, b;
#pragma unroll
        for (int j = 0; j < 4; ++j) {
            a.u[j]     = arow[(kk + hi * 16 + 2 * j)     >> 1];
            a.u[j + 4] = arow[(kk + hi * 16 + 8 + 2 * j) >> 1];
        }
#pragma unroll
        for (int j = 0; j < 8; ++j)
            b.u[j] = brow[(kk + hi * 16 + 2 * j) >> 1];
        acc = __builtin_amdgcn_wmma_f32_16x16x32_f16(
            false, a.v, false, b.v, (short)0, acc, false, false);
    }
    const float bv = bias[l15];
    const int mB = rowBase + hi * 8;
#pragma unroll
    for (int i = 0; i < 8; ++i) {
        float v = acc[i] + bv;
        if (ACT == 1) v = fmaxf(v, 0.0f);
        if (ACT == 2) v = 1.0f / (1.0f + __expf(-v));
        if (l15 < fcols) Yf[(size_t)(mB + i) * ldyf + l15] = v;
    }
}

// ---------------------------------------------------------------------------
// Alpha compositing, one thread per ray (serial cumsum over S samples).
// ---------------------------------------------------------------------------
__global__ void composite_kernel(const float* __restrict__ sigma, const float* __restrict__ rgb,
                                 const float* __restrict__ t, int S,
                                 float* __restrict__ Cout, float* __restrict__ wout) {
    int r = blockIdx.x * blockDim.x + threadIdx.x;
    if (r >= NRAYS) return;
    const float* tr = t     + (size_t)r * S;
    const float* sr = sigma + (size_t)r * S;
    const float* cr = rgb   + (size_t)r * S * 3;
    float acc = 0.0f, C0 = 0.0f, C1 = 0.0f, C2 = 0.0f;
    for (int s = 0; s < S; ++s) {
        float delta = (s < S - 1) ? (tr[s + 1] - tr[s]) : (T_FAR - tr[S - 1]);
        float sd    = sr[s] * delta;
        float Tcur  = __expf(-acc);
        float alpha = 1.0f - __expf(-sd);
        float wv    = Tcur * alpha;
        C0 += wv * cr[s*3 + 0];
        C1 += wv * cr[s*3 + 1];
        C2 += wv * cr[s*3 + 2];
        if (wout) wout[(size_t)r * S + s] = wv;
        acc += sd;
    }
    float bg = __expf(-acc);  // white background
    Cout[r*3 + 0] = C0 + bg;
    Cout[r*3 + 1] = C1 + bg;
    Cout[r*3 + 2] = C2 + bg;
}

// ---------------------------------------------------------------------------
// Inverse-CDF fine sampling + merge/sort with coarse samples.
// One block (256 threads) per ray; bitonic sort of 256 values in LDS.
// ---------------------------------------------------------------------------
__global__ void fine_sample_kernel(const float* __restrict__ t_c, const float* __restrict__ w,
                                   const float* __restrict__ u_f, float* __restrict__ t_all) {
    __shared__ float bins[NC - 1];  // 63
    __shared__ float cdf[NC - 1];   // 63
    __shared__ float tall[256];
    int r = blockIdx.x;
    int tid = threadIdx.x;
    const float* tr = t_c + (size_t)r * NC;
    if (tid == 0) {
        for (int j = 0; j < NC - 1; ++j) bins[j] = 0.5f * (tr[j + 1] + tr[j]);
        float sum = 0.0f;
        for (int j = 0; j < NC - 2; ++j) sum += w[(size_t)r * NC + j + 1] + 1e-5f;
        float c = 0.0f;
        cdf[0] = 0.0f;
        for (int j = 0; j < NC - 2; ++j) {
            c += (w[(size_t)r * NC + j + 1] + 1e-5f) / sum;
            cdf[j + 1] = c;
        }
    }
    __syncthreads();
    if (tid < NC) tall[tid] = tr[tid];
    if (tid < NF) {
        float uu = u_f[(size_t)r * NF + tid];
        int idx = 0;
        for (int k = 0; k < NC - 1; ++k) idx += (cdf[k] <= uu) ? 1 : 0;  // searchsorted right
        int below = idx - 1; if (below < 0) below = 0; if (below > NC - 2) below = NC - 2;
        int above = idx;     if (above < 0) above = 0; if (above > NC - 2) above = NC - 2;
        float cb = cdf[below], ca = cdf[above];
        float bb = bins[below], ba = bins[above];
        float denom = (ca - cb < 1e-5f) ? 1.0f : (ca - cb);
        float frac  = (uu - cb) / denom;
        tall[NC + tid] = bb + frac * (ba - bb);
    }
    if (tid >= SFINE) tall[tid] = 1e30f;  // pad
    __syncthreads();
    for (int k = 2; k <= 256; k <<= 1) {
        for (int j = k >> 1; j > 0; j >>= 1) {
            int ixj = tid ^ j;
            if (ixj > tid) {
                float a = tall[tid], b = tall[ixj];
                bool up = ((tid & k) == 0);
                if ((a > b) == up) { tall[tid] = b; tall[ixj] = a; }
            }
            __syncthreads();
        }
    }
    if (tid < SFINE) t_all[(size_t)r * SFINE + tid] = tall[tid];
}

// ---------------------------------------------------------------------------
// loss = mean((Cc-C)^2) + mean((Cf-C)^2), both means over NRAYS*3.
// ---------------------------------------------------------------------------
__global__ void loss_kernel(const float* __restrict__ Cc, const float* __restrict__ Cf,
                            const float* __restrict__ C, float* __restrict__ out) {
    __shared__ float red[256];
    int tid = threadIdx.x;
    float s = 0.0f;
    for (int i = tid; i < NRAYS * 3; i += 256) {
        float d1 = Cc[i] - C[i];
        float d2 = Cf[i] - C[i];
        s += d1 * d1 + d2 * d2;
    }
    red[tid] = s;
    __syncthreads();
    for (int off = 128; off > 0; off >>= 1) {
        if (tid < off) red[tid] += red[tid + off];
        __syncthreads();
    }
    if (tid == 0) out[0] = red[0] / (float)(NRAYS * 3);
}

// ---------------------------------------------------------------------------
// Host orchestration
// ---------------------------------------------------------------------------
struct MlpW {
    const float *W0, *b0, *W1, *b1, *W2, *b2, *Ws, *bs, *Wf, *bf, *Wr, *br, *Wo, *bo;
};
struct MlpF16 {
    _Float16 *W0, *W1, *W2, *Ws, *Wf, *Wr, *Wo;
    float *b0, *b1, *b2, *bs, *bf, *br, *bo;
};

static void map_params(void* const* d_in, const int* in_sizes, int base, MlpW* m) {
    const float* p[14];
    for (int i = 0; i < 14; ++i) p[i] = (const float*)d_in[base + i];
    bool insertion = (in_sizes[base + 1] == 256);  // b0 follows W0 in insertion order
    if (insertion) {
        m->W0=p[0];  m->b0=p[1];  m->W1=p[2];  m->b1=p[3];  m->W2=p[4];  m->b2=p[5];
        m->Ws=p[6];  m->bs=p[7];  m->Wf=p[8];  m->bf=p[9];  m->Wr=p[10]; m->br=p[11];
        m->Wo=p[12]; m->bo=p[13];
    } else {  // sorted keys: W0,W1,W2,Wf,Wo,Wr,Ws,b0,b1,b2,bf,bo,br,bs
        m->W0=p[0];  m->W1=p[1];  m->W2=p[2];  m->Wf=p[3];  m->Wo=p[4];  m->Wr=p[5];
        m->Ws=p[6];  m->b0=p[7];  m->b1=p[8];  m->b2=p[9];  m->bf=p[10]; m->bo=p[11];
        m->br=p[12]; m->bs=p[13];
    }
}

extern "C" void kernel_launch(void* const* d_in, const int* in_sizes, int n_in,
                              void* d_out, int out_size, void* d_ws, size_t ws_size,
                              hipStream_t stream) {
    (void)n_in; (void)out_size; (void)ws_size;
    const float* o_p  = (const float*)d_in[0];
    const float* d_p  = (const float*)d_in[1];
    const float* C_p  = (const float*)d_in[2];
    const float* uc_p = (const float*)d_in[3];
    const float* uf_p = (const float*)d_in[4];
    MlpW pc, pf;
    map_params(d_in, in_sizes, 5, &pc);
    map_params(d_in, in_sizes, 19, &pf);

    char* base = (char*)d_ws;
    size_t off = 0;
    auto alloc = [&](size_t bytes) -> void* {
        void* p = base + off;
        off += (bytes + 255) & ~(size_t)255;
        return p;
    };
    float* t_c   = (float*)alloc(sizeof(float) * NRAYS * NC);
    float* t_all = (float*)alloc(sizeof(float) * NRAYS * SFINE);
    float* w_c   = (float*)alloc(sizeof(float) * NRAYS * NC);
    float* C_c   = (float*)alloc(sizeof(float) * NRAYS * 3);
    float* C_f   = (float*)alloc(sizeof(float) * NRAYS * 3);
    const int M_FINE = NRAYS * SFINE;  // 196608
    float* sigma = (float*)alloc(sizeof(float) * M_FINE);
    float* rgb   = (float*)alloc(sizeof(float) * M_FINE * 3);

    auto alloc_mlp = [&](MlpF16* m) {
        m->W0 = (_Float16*)alloc(2 * 256 * 64);   m->b0 = (float*)alloc(4 * 256);
        m->W1 = (_Float16*)alloc(2 * 256 * 256);  m->b1 = (float*)alloc(4 * 256);
        m->W2 = (_Float16*)alloc(2 * 256 * 256);  m->b2 = (float*)alloc(4 * 256);
        m->Ws = (_Float16*)alloc(2 * 16 * 256);   m->bs = (float*)alloc(4 * 16);
        m->Wf = (_Float16*)alloc(2 * 256 * 256);  m->bf = (float*)alloc(4 * 256);
        m->Wr = (_Float16*)alloc(2 * 128 * 288);  m->br = (float*)alloc(4 * 128);
        m->Wo = (_Float16*)alloc(2 * 16 * 128);   m->bo = (float*)alloc(4 * 16);
    };
    MlpF16 fc, ff;
    alloc_mlp(&fc);
    alloc_mlp(&ff);

    _Float16* X64 = (_Float16*)alloc(2 * (size_t)CHUNK * 64);
    _Float16* Ha  = (_Float16*)alloc(2 * (size_t)CHUNK * 256);
    _Float16* Hb  = (_Float16*)alloc(2 * (size_t)CHUNK * 256);
    _Float16* Hr  = (_Float16*)alloc(2 * (size_t)CHUNK * 288);
    _Float16* H2  = (_Float16*)alloc(2 * (size_t)CHUNK * 128);

    auto cvt = [&](const float* W, const float* b, int K, int N, int Kpad, int Npad,
                   _Float16* Wt, float* bp) {
        int total = Kpad * Npad;
        cvt_layer<<<(total + 255) / 256, 256, 0, stream>>>(W, b, K, N, Kpad, Npad, Wt, bp);
    };
    auto cvt_mlp = [&](const MlpW& s, const MlpF16& t) {
        cvt(s.W0, s.b0,  63, 256,  64, 256, t.W0, t.b0);
        cvt(s.W1, s.b1, 256, 256, 256, 256, t.W1, t.b1);
        cvt(s.W2, s.b2, 256, 256, 256, 256, t.W2, t.b2);
        cvt(s.Ws, s.bs, 256,   1, 256,  16, t.Ws, t.bs);
        cvt(s.Wf, s.bf, 256, 256, 256, 256, t.Wf, t.bf);
        cvt(s.Wr, s.br, 283, 128, 288, 128, t.Wr, t.br);
        cvt(s.Wo, s.bo, 128,   3, 128,  16, t.Wo, t.bo);
    };
    cvt_mlp(pc, fc);
    cvt_mlp(pf, ff);

    coarse_t_kernel<<<(NRAYS * NC + 255) / 256, 256, 0, stream>>>(uc_p, t_c);

    auto run_mlp = [&](const MlpF16& w, const float* t, int S, int M,
                       float* sig_out, float* rgb_out) {
        for (int p0 = 0; p0 < M; p0 += CHUNK) {
            const int cnt = CHUNK;  // M is a multiple of CHUNK
            pe_points_kernel<<<(cnt + 255) / 256, 256, 0, stream>>>(
                o_p, d_p, t, S, p0, cnt, X64, Hr);
            wmma_gemm_block_kernel<1><<<dim3(cnt / 64, 4), 128, 0, stream>>>(
                X64, 64, w.W0, 64, w.b0, Ha, 256);
            wmma_gemm_block_kernel<1><<<dim3(cnt / 64, 4), 128, 0, stream>>>(
                Ha, 256, w.W1, 256, w.b1, Hb, 256);
            wmma_gemm_block_kernel<1><<<dim3(cnt / 64, 4), 128, 0, stream>>>(
                Hb, 256, w.W2, 256, w.b2, Ha, 256);
            wmma_head_kernel<1><<<dim3(cnt / 16, 1), 32, 0, stream>>>(
                Ha, 256, w.Ws, 256, w.bs, sig_out + p0, 1, 1);
            wmma_gemm_block_kernel<0><<<dim3(cnt / 64, 4), 128, 0, stream>>>(
                Ha, 256, w.Wf, 256, w.bf, Hr, 288);   // feat -> Hr[:, 0:256]
            wmma_gemm_block_kernel<1><<<dim3(cnt / 64, 2), 128, 0, stream>>>(
                Hr, 288, w.Wr, 288, w.br, H2, 128);
            wmma_head_kernel<2><<<dim3(cnt / 16, 1), 32, 0, stream>>>(
                H2, 128, w.Wo, 128, w.bo, rgb_out + (size_t)p0 * 3, 3, 3);
        }
    };

    run_mlp(fc, t_c, NC, NRAYS * NC, sigma, rgb);
    composite_kernel<<<(NRAYS + 255) / 256, 256, 0, stream>>>(sigma, rgb, t_c, NC, C_c, w_c);
    fine_sample_kernel<<<NRAYS, 256, 0, stream>>>(t_c, w_c, uf_p, t_all);
    run_mlp(ff, t_all, SFINE, M_FINE, sigma, rgb);
    composite_kernel<<<(NRAYS + 255) / 256, 256, 0, stream>>>(sigma, rgb, t_all, SFINE, C_f, nullptr);
    loss_kernel<<<1, 256, 0, stream>>>(C_c, C_f, C_p, (float*)d_out);
}
